// RNNEmbedder_42889543418045
// MI455X (gfx1250) — matrix operand order repaired and checked by reading, not compiled, not run on hardware
//
#include <hip/hip_runtime.h>
#include <hip/hip_bf16.h>
#include <math.h>

// ---------------------------------------------------------------------------
// GRU scan, persistent-kernel design for MI455X (gfx1250, wave32, WGP LDS).
//   T=16384, Y=64, H=1024, 3H=3072.
// Phase 1 (optional, ws permitting): ig_all = obs @ Wih^T + bias via
//   v_wmma_f32_16x16x32_bf16 (bf16 A/B, f32 accum).
// Phase 2: persistent scan; W_hh held as bf16 in LDS, 64 workgroups each own
//   16 hidden units (48 rows x 1024). Per-step device-wide barrier via
//   monotonic atomic counter; h double-buffered in global memory. Per-step
//   global reads (ig/obs) are register-staged ahead of the barrier spin and
//   prefetched 8 steps ahead to keep HBM latency off the serial path.
// ---------------------------------------------------------------------------

#define T_STEPS 16384
#define HDIM    1024
#define YDIM    64
#define NWG     64          // workgroups in the persistent scan
#define HPW     16          // hidden units per workgroup (HDIM / NWG)
#define ROWS    48          // 3 gates * HPW rows of W_hh per workgroup
#define PFD     8           // prefetch distance (steps)

typedef __attribute__((ext_vector_type(16))) __bf16 v16bf;
typedef __attribute__((ext_vector_type(2)))  __bf16 bf16x2;
typedef __attribute__((ext_vector_type(8)))  float  v8f;

__device__ __forceinline__ unsigned short f2bf(float f) {
    unsigned u = __float_as_uint(f);
    unsigned r = (u >> 16) & 1u;                 // round-to-nearest-even
    return (unsigned short)((u + 0x7FFFu + r) >> 16);
}
__device__ __forceinline__ __bf16 bits2bf(unsigned short s) {
    return __builtin_bit_cast(__bf16, s);
}
__device__ __forceinline__ float sigmoidf_(float x) {
    return 1.0f / (1.0f + __expf(-x));
}

// --------------------------- init (per-call state) -------------------------
__global__ void init_state(float* gH, unsigned* cnt) {
    if (threadIdx.x == 0) *cnt = 0u;
    for (int k = threadIdx.x; k < 2 * HDIM; k += 256) gH[k] = 0.0f;   // h0 = 0
}

// --------------------------- fp32 -> bf16 convert --------------------------
__global__ void f32_to_bf16(const float* __restrict__ src,
                            unsigned short* __restrict__ dst, int n) {
    int i = blockIdx.x * 256 + threadIdx.x;
    if (i < n) dst[i] = f2bf(src[i]);
}

// ------------------- Phase 1: ig_all = obs @ Wih^T + bias ------------------
// One wave per 16x16 output tile; K = 64 = 2 chunks of 32.
__global__ void __launch_bounds__(256) ig_gemm(
    const unsigned short* __restrict__ A,     // obs  bf16 [16384][64]
    const unsigned short* __restrict__ B,     // Wih  bf16 [3072][64]
    const float* __restrict__ bias,           // [3072]
    float* __restrict__ C)                    // ig_all fp32 [16384][3072]
{
    const int wv   = blockIdx.x * 8 + (threadIdx.x >> 5);
    const int lane = threadIdx.x & 31;
    const int tm   = wv / (3 * HDIM / 16);    // 192 N-tiles
    const int tn   = wv % (3 * HDIM / 16);

    const unsigned short* arow = A + (size_t)(tm * 16 + (lane & 15)) * YDIM;
    const unsigned short* brow = B + (size_t)(tn * 16 + (lane & 15)) * YDIM;
    const int kbA = (lane >> 4) * 8;          // A: two 8-K runs per lane
    const int kbB = (lane >> 4) * 8;          // B: two 8-K runs per lane

    v8f c = {0.f, 0.f, 0.f, 0.f, 0.f, 0.f, 0.f, 0.f};
#pragma unroll
    for (int kc = 0; kc < 2; ++kc) {
        const int k0 = kc * 32;
        v16bf a, b;
#pragma unroll
        for (int i = 0; i < 8; ++i)  a[i]     = bits2bf(arow[k0 + kbA + i]);
#pragma unroll
        for (int i = 0; i < 8; ++i)  a[8 + i] = bits2bf(arow[k0 + 16 + kbA + i]);
#pragma unroll
        for (int i = 0; i < 8; ++i)  b[i]     = bits2bf(brow[k0 + kbB + i]);
#pragma unroll
        for (int i = 0; i < 8; ++i)  b[8 + i] = bits2bf(brow[k0 + 16 + kbB + i]);
        c = __builtin_amdgcn_wmma_f32_16x16x32_bf16(
                /*neg_a=*/false, a, /*neg_b=*/false, b,
                /*c_mod=*/(short)0, c, /*reuse_a=*/false, /*reuse_b=*/false);
    }

    const float bv = bias[tn * 16 + (lane & 15)];
    const int   m0 = tm * 16 + ((lane < 16) ? 0 : 8);
    const int   n  = tn * 16 + (lane & 15);
#pragma unroll
    for (int r = 0; r < 8; ++r)
        C[(size_t)(m0 + r) * (3 * HDIM) + n] = c[r] + bv;
}

// ----------------------------- Phase 2: scan -------------------------------
// IG_GLOBAL=true : read precomputed ig_all (fp32, via L2/HBM, reg-staged)
// IG_GLOBAL=false: compute ig slice inline from obs/Wih (Wih slice in LDS)
template <bool IG_GLOBAL>
__global__ void __launch_bounds__(256, 1) gru_scan(
    const float* __restrict__ obs,      // [T][64]
    const float* __restrict__ Wih,      // [3072][64]
    const float* __restrict__ Whh,      // [3072][1024]
    const float* __restrict__ bias,     // [3072]
    const float* __restrict__ bias_n,   // [1024]
    const float* __restrict__ ig_all,   // [T][3072] (IG_GLOBAL only)
    float* __restrict__ out,            // [T][1024]
    float* gH,                          // [2][1024] double-buffered h
    unsigned* cnt)                      // monotonic barrier counter
{
    __shared__ unsigned sW[ROWS * (HDIM / 2)];     // bf16 pairs: 96 KB
    __shared__ float    sH[HDIM];                  // current h (full vector)
    __shared__ float    sG[ROWS];                  // hg = Whh_slice @ h
    __shared__ float    sIG[ROWS];                 // input-gate preacts
    __shared__ float    sBias[ROWS];
    __shared__ float    sBN[HPW];
    __shared__ float    sWih[IG_GLOBAL ? 1 : ROWS * YDIM];
    __shared__ float    sObs[IG_GLOBAL ? 1 : YDIM];

    const int tid  = threadIdx.x;
    const int lane = tid & 31;
    const int wave = tid >> 5;
    const int j0   = blockIdx.x * HPW;
    // global row owned by threads tid < ROWS (gate-major)
    const int growT = (tid >> 4) * HDIM + j0 + (tid & 15);

    // Preload bf16 W_hh slice: row i -> global row (i/16)*1024 + j0 + (i%16)
    for (int idx = tid; idx < ROWS * (HDIM / 2); idx += 256) {
        const int i = idx >> 9;                  // local row
        const int w = idx & 511;                 // pair index
        const int grow = (i >> 4) * HDIM + j0 + (i & 15);
        const float f0 = Whh[(size_t)grow * HDIM + 2 * w];
        const float f1 = Whh[(size_t)grow * HDIM + 2 * w + 1];
        sW[idx] = (unsigned)f2bf(f0) | ((unsigned)f2bf(f1) << 16);
    }
    if constexpr (!IG_GLOBAL) {
        for (int idx = tid; idx < ROWS * YDIM; idx += 256) {
            const int i = idx >> 6;
            const int k = idx & 63;
            const int grow = (i >> 4) * HDIM + j0 + (i & 15);
            sWih[idx] = Wih[(size_t)grow * YDIM + k];
        }
        if (tid < ROWS) sBias[tid] = bias[growT];
    }
    if (tid < HPW) sBN[tid] = bias_n[j0 + tid];
    for (int k = tid; k < HDIM; k += 256) sH[k] = gH[k];   // h0 (buffer 0)

    // Register-stage step-0 per-step inputs
    float rStage = 0.0f;
    if constexpr (IG_GLOBAL) {
        if (tid < ROWS) rStage = ig_all[growT];
    } else {
        if (tid < YDIM) rStage = obs[tid];
    }
    __syncthreads();

    const float2* sH2 = reinterpret_cast<const float2*>(sH);

    for (int t = 0; t < T_STEPS; ++t) {
        // Commit staged per-step inputs to LDS
        if constexpr (IG_GLOBAL) {
            if (tid < ROWS) sIG[tid] = rStage;
        } else {
            if (tid < YDIM) sObs[tid] = rStage;
        }
        // Prefetch the per-step stream PFD steps ahead (global_prefetch_b8)
        if (t + PFD < T_STEPS) {
            if constexpr (IG_GLOBAL) {
                if (tid < ROWS)
                    __builtin_prefetch(&ig_all[(size_t)(t + PFD) * (3 * HDIM) + growT], 0, 1);
            } else {
                if (tid < YDIM)
                    __builtin_prefetch(&obs[(size_t)(t + PFD) * YDIM + tid], 0, 1);
            }
        }
        __syncthreads();

        // hg = Whh_slice @ h : 6 rows per wave; packed bf16 pair x float2 h
#pragma unroll
        for (int rr = 0; rr < 6; ++rr) {
            const int i = wave * 6 + rr;
            const unsigned* wrow = &sW[i * (HDIM / 2)];
            float acc = 0.0f;
#pragma unroll
            for (int it = 0; it < 16; ++it) {
                const bf16x2 wp = __builtin_bit_cast(bf16x2, wrow[lane + 32 * it]);
                const float2 hv = sH2[lane + 32 * it];   // ds_load_b64, no bank conflict
                acc = fmaf((float)wp.x, hv.x, acc);      // v_fma_mix_f32_bf16 candidates
                acc = fmaf((float)wp.y, hv.y, acc);
            }
#pragma unroll
            for (int off = 16; off; off >>= 1) acc += __shfl_xor(acc, off, 32);
            if (lane == 0) sG[i] = acc;

            if constexpr (!IG_GLOBAL) {     // inline ig for this row (K=64)
                float ia = sWih[i * YDIM + lane]      * sObs[lane]
                         + sWih[i * YDIM + 32 + lane] * sObs[32 + lane];
#pragma unroll
                for (int off = 16; off; off >>= 1) ia += __shfl_xor(ia, off, 32);
                if (lane == 0) sIG[i] = ia + sBias[i];
            }
        }
        __syncthreads();

        // Gates + state update for this WG's 16 hidden units
        if (tid < HPW) {
            const int h = j0 + tid;
            const float r = sigmoidf_(sIG[tid] + sG[tid]);
            const float z = sigmoidf_(sIG[HPW + tid] + sG[HPW + tid]);
            const float n = tanhf(sIG[2 * HPW + tid] +
                                  r * (sG[2 * HPW + tid] + sBN[tid]));
            const float hn = n + z * (sH[h] - n);
            out[(size_t)t * HDIM + h] = hn;
            __hip_atomic_store(&gH[((t + 1) & 1) * HDIM + h], hn,
                               __ATOMIC_RELAXED, __HIP_MEMORY_SCOPE_AGENT);
            __threadfence();                 // agent-visible before barrier
        }
        __syncthreads();

        if (t + 1 < T_STEPS) {
            // Issue next step's global read BEFORE the spin so its latency
            // overlaps the device-wide barrier wait.
            if constexpr (IG_GLOBAL) {
                if (tid < ROWS)
                    rStage = ig_all[(size_t)(t + 1) * (3 * HDIM) + growT];
            } else {
                if (tid < YDIM)
                    rStage = obs[(size_t)(t + 1) * YDIM + tid];
            }
            if (tid == 0)
                __hip_atomic_fetch_add(cnt, 1u, __ATOMIC_RELEASE,
                                       __HIP_MEMORY_SCOPE_AGENT);
            const unsigned target = (unsigned)NWG * (unsigned)(t + 1);
            while (__hip_atomic_load(cnt, __ATOMIC_ACQUIRE,
                                     __HIP_MEMORY_SCOPE_AGENT) < target) {
                __builtin_amdgcn_s_sleep(1);
            }
            const float* src = &gH[((t + 1) & 1) * HDIM];
            for (int k = tid; k < HDIM; k += 256) sH[k] = src[k];
            __syncthreads();
        }
    }
}

// ------------------------------- launcher ----------------------------------
extern "C" void kernel_launch(void* const* d_in, const int* in_sizes, int n_in,
                              void* d_out, int out_size, void* d_ws, size_t ws_size,
                              hipStream_t stream) {
    const float* obs    = (const float*)d_in[0];   // [16384,64]
    const float* wih    = (const float*)d_in[1];   // [3072,64]
    const float* whh    = (const float*)d_in[2];   // [3072,1024]
    const float* bias   = (const float*)d_in[3];   // [3072]
    const float* bias_n = (const float*)d_in[4];   // [1024]
    float* out = (float*)d_out;

    char* ws = (char*)d_ws;
    unsigned*       cnt  = (unsigned*)ws;                         // @0
    float*          gH   = (float*)(ws + 256);                    // 8 KB
    unsigned short* obsb = (unsigned short*)(ws + 16384);         // 2 MB
    unsigned short* wihb = (unsigned short*)(ws + 2113536);       // 384 KB
    float*          ig   = (float*)(ws + 4194304);                // 192 MB

    const size_t need = 4194304ull + (size_t)T_STEPS * 3 * HDIM * sizeof(float);

    init_state<<<1, 256, 0, stream>>>(gH, cnt);

    if (ws_size >= need) {
        f32_to_bf16<<<(T_STEPS * YDIM + 255) / 256, 256, 0, stream>>>(
            obs, obsb, T_STEPS * YDIM);
        f32_to_bf16<<<(3 * HDIM * YDIM + 255) / 256, 256, 0, stream>>>(
            wih, wihb, 3 * HDIM * YDIM);
        ig_gemm<<<(T_STEPS / 16) * (3 * HDIM / 16) / 8, 256, 0, stream>>>(
            obsb, wihb, bias, ig);
        gru_scan<true><<<NWG, 256, 0, stream>>>(
            obs, wih, whh, bias, bias_n, ig, out, gH, cnt);
    } else {
        gru_scan<false><<<NWG, 256, 0, stream>>>(
            obs, wih, whh, bias, bias_n, nullptr, out, gH, cnt);
    }
}
